// LSTM_2903397893016
// MI455X (gfx1250) — matrix operand order repaired and checked by reading, not compiled, not run on hardware
//
#include <hip/hip_runtime.h>

typedef __attribute__((ext_vector_type(16))) __bf16 v16bf;
typedef __attribute__((ext_vector_type(16))) float  v16f;
typedef __attribute__((ext_vector_type(8)))  float  v8f;
typedef __attribute__((__vector_size__(16))) int    v4i;

#define Bsz 16
#define Tsz 4096
#define Dsz 256
#define Msz (Bsz * Tsz)     /* 65536 rows */
#define Ksz 768             /* 512 (input|h) + 256 (c) */
#define NG  1024            /* 4 gates * 256 */
#define M_TILE 128
#define D_TILE 32
#define K_STEP 32
#define A_STRIDE 80         /* bytes per LDS row: 64B data + 16B pad, 16B aligned */
#define B_STRIDE 80
#define LDS_A_BYTES (M_TILE * A_STRIDE)   /* 10240 */
#define LDS_B_BYTES (128 * B_STRIDE)      /* 10240 */

#if __has_builtin(__builtin_amdgcn_global_load_async_to_lds_b128) && \
    __has_builtin(__builtin_amdgcn_s_wait_asynccnt)
#define HAS_ASYNC_LDS 1
#else
#define HAS_ASYNC_LDS 0
#endif

#define AS1 __attribute__((address_space(1)))
#define AS3 __attribute__((address_space(3)))

__device__ __forceinline__ float sigmoidf_fast(float x) {
    return 1.0f / (1.0f + __expf(-x));
}

// ---------------------------------------------------------------------------
// Pack the 8 f32 weight matrices into one bf16 buffer Wp[gc][k], k contiguous.
// gc = g*256 + d;  k<512 -> W_x[g][k][d], k>=512 -> W_h[g][k-512][d]
// ---------------------------------------------------------------------------
__global__ __launch_bounds__(256) void lstm_pack_weights(
    const float* __restrict__ Wii, const float* __restrict__ Whi,
    const float* __restrict__ Wif, const float* __restrict__ Whf,
    const float* __restrict__ Wig, const float* __restrict__ Whg,
    const float* __restrict__ Wio, const float* __restrict__ Who,
    unsigned short* __restrict__ Wp)
{
    int idx = blockIdx.x * 256 + threadIdx.x;
    if (idx >= NG * Ksz) return;
    int gc = idx / Ksz;
    int k  = idx - gc * Ksz;
    int g  = gc >> 8;
    int d  = gc & 255;
    const float* W;
    int rk;
    if (k < 512) {
        W  = (g == 0) ? Wii : (g == 1) ? Wif : (g == 2) ? Wig : Wio;
        rk = k;
    } else {
        W  = (g == 0) ? Whi : (g == 1) ? Whf : (g == 2) ? Whg : Who;
        rk = k - 512;
    }
    __bf16 b = (__bf16)W[(size_t)rk * Dsz + d];   // RNE trunc
    Wp[idx] = __builtin_bit_cast(unsigned short, b);
}

// ---------------------------------------------------------------------------
// Fused GEMM + LSTM gate epilogue.
// Workgroup: 256 threads (8 wave32), tile = 128 rows x 32 d-cols (x 4 gates).
// Wave (wM 0..3, wN 0..1): 32 rows x 16 d-cols x 4 gates -> acc[2][4] (v8f).
// ---------------------------------------------------------------------------
union FragBF { uint4 u[2]; v16bf v; };

__global__ __launch_bounds__(256) void lstm_wmma_kernel(
    const float* __restrict__ input, const float* __restrict__ h,
    const float* __restrict__ c, const unsigned short* __restrict__ Wp,
    float* __restrict__ out)
{
    __shared__ __align__(16) unsigned char lds[LDS_A_BYTES + LDS_B_BYTES];
    unsigned char* ldsA = lds;
    unsigned char* ldsB = lds + LDS_A_BYTES;

    const int tid    = threadIdx.x;
    const int lane   = tid & 31;
    const int laneLo = lane & 15;
    const int laneHi = lane >> 4;
    const int wave   = tid >> 5;
    const int waveM  = wave & 3;   // row group 0..3 (32 rows each)
    const int waveN  = wave >> 2;  // d-col group 0..1 (16 cols each)

    const int rowBase = blockIdx.x * M_TILE;
    const int dBase   = blockIdx.y * D_TILE;

    // Cooperative fill indices (each thread moves 16 elements / step)
    const int am = tid >> 1;              // tile row 0..127
    const int ak = (tid & 1) * 16;        // k sub-offset {0,16}
    const int bcol = tid >> 1;            // tile col 0..127  (g*32 + dloc)
    const int bg   = bcol >> 5;
    const int bd   = bcol & 31;
    const unsigned short* wbase =
        Wp + (size_t)(bg * 256 + dBase + bd) * Ksz + ak;
    const size_t arow = (size_t)(rowBase + am);
    unsigned char* bdst = ldsB + bcol * B_STRIDE + ak * 2;

    v8f acc[2][4];
    const v8f vzero = {0.f, 0.f, 0.f, 0.f, 0.f, 0.f, 0.f, 0.f};
#pragma unroll
    for (int mt = 0; mt < 2; ++mt)
#pragma unroll
        for (int g = 0; g < 4; ++g) acc[mt][g] = vzero;

    for (int step = 0; step < Ksz / K_STEP; ++step) {
        const int kk = step * K_STEP;
        const float* src; int col0;
        if (kk < 256)      { src = input; col0 = kk;       }
        else if (kk < 512) { src = h;     col0 = kk - 256; }
        else               { src = c;     col0 = kk - 512; }

        // A-operand global loads for this step (issued before the barrier so
        // they overlap the previous step's WMMAs)
        const float4* ap = (const float4*)(src + arow * Dsz + col0 + ak);
        float4 f0 = ap[0], f1 = ap[1], f2 = ap[2], f3 = ap[3];
#if !HAS_ASYNC_LDS
        const uint4* wp4 = (const uint4*)(wbase + kk);
        uint4 w0 = wp4[0], w1 = wp4[1];
#endif

        // Prefetch next step's activation chunk
        const int kn = kk + K_STEP;
        if (kn < Ksz) {
            const float* ns; int nc;
            if (kn < 256)      { ns = input; nc = kn;       }
            else if (kn < 512) { ns = h;     nc = kn - 256; }
            else               { ns = c;     nc = kn - 512; }
            __builtin_prefetch(ns + arow * Dsz + nc + ak, 0, 0);
        }

        __syncthreads();   // all waves done reading LDS from previous step

#if HAS_ASYNC_LDS
        // B-operand: async DMA global(bf16,L2-resident) -> LDS, no VGPR trip.
        __builtin_amdgcn_global_load_async_to_lds_b128(
            (AS1 v4i*)(wbase + kk), (AS3 v4i*)bdst, 0, 0);
        __builtin_amdgcn_global_load_async_to_lds_b128(
            (AS1 v4i*)(wbase + kk), (AS3 v4i*)bdst, 16, 0);
#endif

        // Convert A to bf16 (compiler picks native cvt; RNE) and stage
        v16f fav = { f0.x, f0.y, f0.z, f0.w, f1.x, f1.y, f1.z, f1.w,
                     f2.x, f2.y, f2.z, f2.w, f3.x, f3.y, f3.z, f3.w };
        FragBF ab;
        ab.v = __builtin_convertvector(fav, v16bf);
        unsigned char* adst = ldsA + am * A_STRIDE + ak * 2;
        *(uint4*)(adst)      = ab.u[0];
        *(uint4*)(adst + 16) = ab.u[1];
#if !HAS_ASYNC_LDS
        *(uint4*)(bdst)      = w0;
        *(uint4*)(bdst + 16) = w1;
#endif

#if HAS_ASYNC_LDS
        __builtin_amdgcn_s_wait_asynccnt(0);
#endif
        __syncthreads();

        // A fragments (ISA 7.12.2, 16-bit A 16x32):
        // lanes 0-15 hold K0-7 (bytes 0..15) + K16-23 (bytes 32..47)
        // lanes 16-31 hold K8-15 (16..31) + K24-31 (48..63)
        FragBF afrag[2];
#pragma unroll
        for (int mt = 0; mt < 2; ++mt) {
            const int row = waveM * 32 + mt * 16 + laneLo;
            const unsigned char* p = ldsA + row * A_STRIDE + laneHi * 16;
            afrag[mt].u[0] = *(const uint4*)p;
            afrag[mt].u[1] = *(const uint4*)(p + 32);
        }
        // B fragments (32x16): lanes 0-15 hold K0-7 + K16-23, lanes 16-31 K8-15 + K24-31,
        // column = lane%16; LDS B is column-major (K contiguous per column).
#pragma unroll
        for (int g = 0; g < 4; ++g) {
            const int colT = g * 32 + waveN * 16 + laneLo;
            const unsigned char* p = ldsB + colT * B_STRIDE + laneHi * 16;
            FragBF bfrag;
            bfrag.u[0] = *(const uint4*)p;
            bfrag.u[1] = *(const uint4*)(p + 32);
#pragma unroll
            for (int mt = 0; mt < 2; ++mt) {
                acc[mt][g] = __builtin_amdgcn_wmma_f32_16x16x32_bf16(
                    false, afrag[mt].v, false, bfrag.v,
                    (short)0, acc[mt][g], false, false);
            }
        }
    }

    // Epilogue: gates align lane-for-lane across the 4 accumulators.
    // C/D layout: VGPR v, lanes 0-15 -> M=v, N=lane; lanes 16-31 -> M=v+8.
    const int d = dBase + waveN * 16 + laneLo;
    const size_t HN = (size_t)Msz * Dsz;   // offset of c_new in d_out
#pragma unroll
    for (int mt = 0; mt < 2; ++mt) {
#pragma unroll
        for (int v = 0; v < 8; ++v) {
            const size_t row =
                (size_t)(rowBase + waveM * 32 + mt * 16 + v + 8 * laneHi);
            const float zi = acc[mt][0][v];
            const float zf = acc[mt][1][v];
            const float zg = acc[mt][2][v];
            const float zo = acc[mt][3][v];
            const float cold = c[row * Dsz + d];
            const float ig = sigmoidf_fast(zi);
            const float fg = sigmoidf_fast(zf);
            const float gg = tanhf(zg);
            const float og = sigmoidf_fast(zo);
            const float cn = fg * cold + ig * gg;
            const float hn = og * tanhf(cn);
            out[row * Dsz + d]      = hn;
            out[HN + row * Dsz + d] = cn;
        }
    }
}

// ---------------------------------------------------------------------------
extern "C" void kernel_launch(void* const* d_in, const int* in_sizes, int n_in,
                              void* d_out, int out_size, void* d_ws, size_t ws_size,
                              hipStream_t stream) {
    (void)in_sizes; (void)n_in; (void)out_size; (void)ws_size;
    const float* input = (const float*)d_in[0];
    // d_in[1] = input2 : unused by the reference
    const float* h   = (const float*)d_in[2];
    const float* c   = (const float*)d_in[3];
    const float* Wii = (const float*)d_in[4];
    const float* Whi = (const float*)d_in[5];
    const float* Wif = (const float*)d_in[6];
    const float* Whf = (const float*)d_in[7];
    const float* Wig = (const float*)d_in[8];
    const float* Whg = (const float*)d_in[9];
    const float* Wio = (const float*)d_in[10];
    const float* Who = (const float*)d_in[11];

    unsigned short* Wp = (unsigned short*)d_ws;   // 1024*768 bf16 = 1.5 MB
    float* out = (float*)d_out;

    const int packElems  = NG * Ksz;
    const int packBlocks = (packElems + 255) / 256;
    lstm_pack_weights<<<packBlocks, 256, 0, stream>>>(
        Wii, Whi, Wif, Whf, Wig, Whg, Wio, Who, Wp);

    dim3 grid(Msz / M_TILE, Dsz / D_TILE);   // 512 x 8
    lstm_wmma_kernel<<<grid, 256, 0, stream>>>(input, h, c, Wp, out);
}
